// HeteroLinkGCN_57543971832738
// MI455X (gfx1250) — matrix operand order repaired and checked by reading, not compiled, not run on hardware
//
#include <hip/hip_runtime.h>
#include <hip/hip_bf16.h>

// ---------------------------------------------------------------------------
// HeteroLinkGCN on MI455X (gfx1250):
//  * scatter-mean aggregation: bandwidth/atomic bound, accumulators L2-resident
//  * GEMMs: v_wmma_f32_16x16x32_bf16 against one-time bf16-transposed weights
//    (B fragments = two 8-bf16 runs of a W^T row, L1-resident), 64 rows/block
//    so every B fragment feeds 4 independent WMMAs.
// ---------------------------------------------------------------------------

typedef __attribute__((ext_vector_type(16))) __bf16 v16bf;
typedef __attribute__((ext_vector_type(8)))  __bf16 v8bf;
typedef __attribute__((ext_vector_type(8)))  float  v8f;

#define D 128         // feature dim everywhere
#define ROWS 64       // output rows per GEMM block (4 row-tiles of 16)

// ------------------------------ utility ------------------------------------

__global__ void fill_zero_kernel(float* __restrict__ p, size_t n) {
    size_t i = (size_t)blockIdx.x * blockDim.x + threadIdx.x;
    size_t stride = (size_t)gridDim.x * blockDim.x;
    for (; i < n; i += stride) p[i] = 0.0f;
}

__global__ void degree_kernel(const int* __restrict__ dst, float* __restrict__ deg, int E) {
    int i = blockIdx.x * blockDim.x + threadIdx.x;
    if (i < E) atomicAdd(&deg[dst[i]], 1.0f);
}

// One wave32 per edge: gather 128-f32 source row as float4/lane, scale by edge
// weight, atomic-add into destination accumulator (stays in 192MB L2).
__global__ __launch_bounds__(256) void scatter_kernel(
    const float* __restrict__ xsrc, const float* __restrict__ w,
    const int* __restrict__ src, const int* __restrict__ dst,
    float* __restrict__ acc, int E) {
    int edge = blockIdx.x * 8 + (threadIdx.x >> 5);
    int lane = threadIdx.x & 31;
    if (edge >= E) return;
    int s = src[edge], d = dst[edge];
    float we = w[edge];
    const float4* srow = (const float4*)(xsrc + (size_t)s * D);
    float4 v = srow[lane];
    float* drow = acc + (size_t)d * D + lane * 4;
    atomicAdd(drow + 0, v.x * we);
    atomicAdd(drow + 1, v.y * we);
    atomicAdd(drow + 2, v.z * we);
    atomicAdd(drow + 3, v.w * we);
}

// neigh[row, :] /= max(deg[row], 1)
__global__ void normalize_kernel(float* __restrict__ acc, const float* __restrict__ deg, int n) {
    int i = blockIdx.x * blockDim.x + threadIdx.x;
    if (i >= n) return;
    float d = deg[i >> 7];
    acc[i] = acc[i] / fmaxf(d, 1.0f);
}

// One-time: W[k][n] (f32, 128x128) -> Wt[n][k] (bf16).  16384 threads total.
__global__ void transpose_w_bf16_kernel(const float* __restrict__ W, __bf16* __restrict__ Wt) {
    int i = blockIdx.x * blockDim.x + threadIdx.x;   // coalesced read of W
    int k = i >> 7, n = i & (D - 1);
    Wt[n * D + k] = (__bf16)W[i];
}

// ------------------------------ WMMA GEMM ----------------------------------
// out[M,128] = epilogue( scale * ( sum_p A_p @ B_p  + bias0 + bias1 ) )
// Block: 64 rows x 128 cols, 256 threads. Wave w owns column tile [16w,16w+16)
// and 4 row-tiles -> 4 f32 accumulators. K=128 -> 4 K-tiles; per K-tile one
// B fragment (2x16B loads from bf16 W^T) feeds 4 v_wmma_f32_16x16x32_bf16.

struct GemmArgs {
    const float*  A0; const float*  A1; const float*  A2; const float*  A3;
    const __bf16* B0; const __bf16* B1; const __bf16* B2; const __bf16* B3;  // W^T, bf16
    const float* bias0; const float* bias1;   // bias1 may be null
    float scale;
    int npairs;
    int relu;
    int M;
    float* out;
};

__global__ __launch_bounds__(256) void gemm_sage_kernel(GemmArgs g) {
    const int lane  = threadIdx.x & 31;
    const int wave  = threadIdx.x >> 5;
    const int rbase = blockIdx.x * ROWS;
    const int col   = wave * 16 + (lane & 15);           // output column (N)
    const int hi    = (lane >> 4) & 1;                   // lane group 16..31?

    const float*  As[4] = {g.A0, g.A1, g.A2, g.A3};
    const __bf16* Bs[4] = {g.B0, g.B1, g.B2, g.B3};

    // Bias depends only on N -> same value in every accumulator element.
    float bval = g.bias0[col] + (g.bias1 ? g.bias1[col] : 0.0f);
    v8f c[4];
#pragma unroll
    for (int t = 0; t < 4; ++t)
#pragma unroll
        for (int r = 0; r < 8; ++r) c[t][r] = bval;

    const int kbA = (lane & 16) ? 8 : 0;    // A: lanes 16-31 hold K offset +8
    const int kbB = (lane & 16) ? 8 : 0;    // B: lanes 16-31 hold K offset +8

    for (int p = 0; p < g.npairs; ++p) {
        const __bf16* Wt = Bs[p];
        const float* Arow[4];
#pragma unroll
        for (int t = 0; t < 4; ++t) {
            int r = rbase + 16 * t + (lane & 15);
            if (r >= g.M) r = g.M - 1;                   // clamp, no divergence
            Arow[t] = As[p] + (size_t)r * D;
        }

#pragma unroll
        for (int k0 = 0; k0 < D; k0 += 32) {
            // B fragment: bf16 of W^T row 'col' at K kbB..kbB+7 and kbB+16..kbB+23 -> 2x 16B loads.
            const __bf16* bp = Wt + (size_t)col * D + k0 + kbB;
            union { v16bf v; v8bf h[2]; } bu;
            bu.h[0] = *(const v8bf*)(bp);
            bu.h[1] = *(const v8bf*)(bp + 16);

#pragma unroll
            for (int t = 0; t < 4; ++t) {
                // A fragment (ISA 7.12.2, 16-bit A 16x32): VGPR v<4 -> K=kbA+2v,
                // v>=4 -> K=kbA+16+2(v-4); 2 consecutive K -> float2 load.
                v16bf a;
#pragma unroll
                for (int v = 0; v < 8; ++v) {
                    int ka = k0 + kbA + ((v < 4) ? (2 * v) : (16 + 2 * (v - 4)));
                    float2 av = *(const float2*)(Arow[t] + ka);
                    a[2 * v]     = (__bf16)av.x;
                    a[2 * v + 1] = (__bf16)av.y;
                }
                c[t] = __builtin_amdgcn_wmma_f32_16x16x32_bf16(
                           /*neg_a=*/false, a, /*neg_b=*/false, bu.v,
                           /*c_mod=*/(short)0, c[t], /*reuse_a=*/false, /*reuse_b=*/false);
            }
        }
    }

    // Epilogue: C/D layout — VGPR r, lane group hi -> row = r + 8*hi.
#pragma unroll
    for (int t = 0; t < 4; ++t) {
        int rt = rbase + 16 * t + 8 * hi;
#pragma unroll
        for (int r = 0; r < 8; ++r) {
            int rr = rt + r;
            float v = c[t][r] * g.scale;
            if (g.relu) v = fmaxf(v, 0.0f);
            if (rr < g.M) g.out[(size_t)rr * D + col] = v;
        }
    }
}

// ------------------------------ launcher -----------------------------------

static inline void launch_gemm(const float* A0, const __bf16* B0,
                               const float* A1, const __bf16* B1,
                               const float* A2, const __bf16* B2,
                               const float* A3, const __bf16* B3,
                               const float* bias0, const float* bias1,
                               float scale, int npairs, int relu, int M,
                               float* out, hipStream_t stream) {
    GemmArgs g;
    g.A0 = A0; g.A1 = A1 ? A1 : A0; g.A2 = A2 ? A2 : A0; g.A3 = A3 ? A3 : A0;
    g.B0 = B0; g.B1 = B1 ? B1 : B0; g.B2 = B2 ? B2 : B0; g.B3 = B3 ? B3 : B0;
    g.bias0 = bias0; g.bias1 = bias1;
    g.scale = scale; g.npairs = npairs; g.relu = relu; g.M = M; g.out = out;
    int blocks = (M + ROWS - 1) / ROWS;
    gemm_sage_kernel<<<blocks, 256, 0, stream>>>(g);
}

extern "C" void kernel_launch(void* const* d_in, const int* in_sizes, int n_in,
                              void* d_out, int out_size, void* d_ws, size_t ws_size,
                              hipStream_t stream) {
    // ---- inputs (setup_inputs order) ----
    const float* x_ac   = (const float*)d_in[0];
    const float* x_w    = (const float*)d_in[1];
    const float* w_tic  = (const float*)d_in[2];
    const float* w_w    = (const float*)d_in[3];
    const float* w_rel  = (const float*)d_in[4];
    const int*   src_tic = (const int*)d_in[5];
    const int*   dst_tic = (const int*)d_in[6];
    const int*   src_w   = (const int*)d_in[7];
    const int*   dst_w   = (const int*)d_in[8];
    const int*   src_rel = (const int*)d_in[9];
    const int*   dst_rel = (const int*)d_in[10];
    // per layer, per edge-type: Ws, Wn, b  (indices 11..28)
    const float* Wmat[12] = {
        (const float*)d_in[11], (const float*)d_in[12],   // l1 tic Ws, Wn
        (const float*)d_in[14], (const float*)d_in[15],   // l1 w   Ws, Wn
        (const float*)d_in[17], (const float*)d_in[18],   // l1 rel Ws, Wn
        (const float*)d_in[20], (const float*)d_in[21],   // l2 tic Ws, Wn
        (const float*)d_in[23], (const float*)d_in[24],   // l2 w   Ws, Wn
        (const float*)d_in[26], (const float*)d_in[27],   // l2 rel Ws, Wn
    };
    const float* l1_tic_b = (const float*)d_in[13];
    const float* l1_w_b   = (const float*)d_in[16];
    const float* l1_rel_b = (const float*)d_in[19];
    const float* l2_tic_b = (const float*)d_in[22];
    const float* l2_w_b   = (const float*)d_in[25];
    const float* l2_rel_b = (const float*)d_in[28];

    const int Nac  = in_sizes[0] / D;   // 50000
    const int Nw   = in_sizes[1] / D;   // 20000
    const int Etic = in_sizes[2];       // 800000
    const int Ew   = in_sizes[3];       // 320000
    const int Erel = in_sizes[4];       // 400000

    // ---- workspace layout ----
    float* ws       = (float*)d_ws;
    float* neigh_ac = ws;                                  // Nac*D
    float* neigh_w1 = neigh_ac + (size_t)Nac * D;          // Nw*D
    float* neigh_w2 = neigh_w1 + (size_t)Nw * D;           // Nw*D
    float* h1_ac    = neigh_w2 + (size_t)Nw * D;           // Nac*D
    float* h1_w     = h1_ac + (size_t)Nac * D;             // Nw*D
    float* deg_ac   = h1_w + (size_t)Nw * D;               // Nac
    float* deg_w1   = deg_ac + Nac;                        // Nw
    float* deg_w2   = deg_w1 + Nw;                         // Nw
    // bf16 W^T region (16B-aligned: preceding float count is even)
    __bf16* wt_base = (__bf16*)(deg_w2 + Nw);
    __bf16* Wt[12];
    for (int m = 0; m < 12; ++m) Wt[m] = wt_base + (size_t)m * D * D;

    const size_t neigh_total = (size_t)(Nac + 2 * Nw) * D;
    const size_t deg_total   = (size_t)Nac + 2 * Nw;

    float* out_ac = (float*)d_out;
    float* out_w  = out_ac + (size_t)Nac * D;

    // ---- one-time: bf16-transpose all 12 weight matrices ----
    for (int m = 0; m < 12; ++m)
        transpose_w_bf16_kernel<<<(D * D) / 256, 256, 0, stream>>>(Wmat[m], Wt[m]);

    // ---- degrees (layer-invariant) ----
    fill_zero_kernel<<<2048, 256, 0, stream>>>(deg_ac, deg_total);
    degree_kernel<<<(Etic + 255) / 256, 256, 0, stream>>>(dst_tic, deg_ac, Etic);
    degree_kernel<<<(Ew + 255) / 256, 256, 0, stream>>>(dst_w, deg_w1, Ew);
    degree_kernel<<<(Erel + 255) / 256, 256, 0, stream>>>(dst_rel, deg_w2, Erel);

    // ================= layer 1 =================
    fill_zero_kernel<<<4096, 256, 0, stream>>>(neigh_ac, neigh_total);
    scatter_kernel<<<(Etic + 7) / 8, 256, 0, stream>>>(x_ac, w_tic, src_tic, dst_tic, neigh_ac, Etic);
    scatter_kernel<<<(Ew + 7) / 8, 256, 0, stream>>>(x_w, w_w, src_w, dst_w, neigh_w1, Ew);
    scatter_kernel<<<(Erel + 7) / 8, 256, 0, stream>>>(x_ac, w_rel, src_rel, dst_rel, neigh_w2, Erel);
    normalize_kernel<<<(Nac * D + 255) / 256, 256, 0, stream>>>(neigh_ac, deg_ac, Nac * D);
    normalize_kernel<<<(Nw * D + 255) / 256, 256, 0, stream>>>(neigh_w1, deg_w1, Nw * D);
    normalize_kernel<<<(Nw * D + 255) / 256, 256, 0, stream>>>(neigh_w2, deg_w2, Nw * D);

    // h1_ac = relu(x_ac @ tic_Ws + tic_b + neigh_ac @ tic_Wn)
    launch_gemm(x_ac, Wt[0], neigh_ac, Wt[1], nullptr, nullptr, nullptr, nullptr,
                l1_tic_b, nullptr, 1.0f, 2, 1, Nac, h1_ac, stream);
    // h1_w = relu(0.5*(x_w@w_Ws + neigh_w1@w_Wn + x_w@rel_Ws + neigh_w2@rel_Wn + w_b + rel_b))
    launch_gemm(x_w, Wt[2], neigh_w1, Wt[3], x_w, Wt[4], neigh_w2, Wt[5],
                l1_w_b, l1_rel_b, 0.5f, 4, 1, Nw, h1_w, stream);

    // ================= layer 2 =================
    fill_zero_kernel<<<4096, 256, 0, stream>>>(neigh_ac, neigh_total);
    scatter_kernel<<<(Etic + 7) / 8, 256, 0, stream>>>(h1_ac, w_tic, src_tic, dst_tic, neigh_ac, Etic);
    scatter_kernel<<<(Ew + 7) / 8, 256, 0, stream>>>(h1_w, w_w, src_w, dst_w, neigh_w1, Ew);
    scatter_kernel<<<(Erel + 7) / 8, 256, 0, stream>>>(h1_ac, w_rel, src_rel, dst_rel, neigh_w2, Erel);
    normalize_kernel<<<(Nac * D + 255) / 256, 256, 0, stream>>>(neigh_ac, deg_ac, Nac * D);
    normalize_kernel<<<(Nw * D + 255) / 256, 256, 0, stream>>>(neigh_w1, deg_w1, Nw * D);
    normalize_kernel<<<(Nw * D + 255) / 256, 256, 0, stream>>>(neigh_w2, deg_w2, Nw * D);

    launch_gemm(h1_ac, Wt[6], neigh_ac, Wt[7], nullptr, nullptr, nullptr, nullptr,
                l2_tic_b, nullptr, 1.0f, 2, 0, Nac, out_ac, stream);
    launch_gemm(h1_w, Wt[8], neigh_w1, Wt[9], h1_w, Wt[10], neigh_w2, Wt[11],
                l2_w_b, l2_rel_b, 0.5f, 4, 0, Nw, out_w, stream);
}
